// SeparableCLIPDecoderBlock_31868657336834
// MI455X (gfx1250) — matrix operand order repaired and checked by reading, not compiled, not run on hardware
//
#include <hip/hip_runtime.h>

typedef _Float16 half_t;
typedef __attribute__((ext_vector_type(16))) _Float16 v16h;
typedef __attribute__((ext_vector_type(8)))  float    v8f;

#define BATCH 8
#define HW    4096   // 64*64

// -------- CDNA5 async global->LDS support (guarded; falls back to sync copy) --
#if defined(__gfx1250__) && __has_builtin(__builtin_amdgcn_global_load_async_to_lds_b128)
#define ASYNC_LDS 1
typedef int v4i_t __attribute__((__vector_size__(16)));
typedef __attribute__((address_space(1))) void  gvoid_t;
typedef __attribute__((address_space(3))) void  svoid_t;
typedef __attribute__((address_space(1))) v4i_t gv4i_t;
typedef __attribute__((address_space(3))) v4i_t sv4i_t;
#endif

__device__ __forceinline__ void cp16_g2l(half_t* l, const half_t* g) {
#if defined(ASYNC_LDS)
  __builtin_amdgcn_global_load_async_to_lds_b128(
      (gv4i_t*)(gvoid_t*)g, (sv4i_t*)(svoid_t*)l, 0, 0);
#else
  *(uint4*)l = *(const uint4*)g;
#endif
}

template <int N>
__device__ __forceinline__ void wait_async_le() {
#if defined(ASYNC_LDS)
#if __has_builtin(__builtin_amdgcn_s_wait_asynccnt)
  __builtin_amdgcn_s_wait_asynccnt(N);
#else
  asm volatile("s_wait_asynccnt %0" ::"n"(N) : "memory");
#endif
#endif
}

// ---------------------------------------------------------------------------
// Bilinear upsample x2, align_corners=True: 32x32 -> 64x64 (NCHW f32)
// ---------------------------------------------------------------------------
__global__ __launch_bounds__(256)
void k_upsample_ac(const float* __restrict__ src, float* __restrict__ dst) {
  int i = blockIdx.x * 256 + threadIdx.x;          // over B*C*64*64
  int x = i & 63, y = (i >> 6) & 63;
  int bc = i >> 12;
  const float s = 31.0f / 63.0f;
  float fy = y * s, fx = x * s;
  int y0 = (int)fy, x0 = (int)fx;
  int y1 = y0 + 1 > 31 ? 31 : y0 + 1;
  int x1 = x0 + 1 > 31 ? 31 : x0 + 1;
  float ty = fy - (float)y0, tx = fx - (float)x0;
  const float* p = src + (size_t)bc * 1024;
  float v00 = p[y0 * 32 + x0], v01 = p[y0 * 32 + x1];
  float v10 = p[y1 * 32 + x0], v11 = p[y1 * 32 + x1];
  float v0 = v00 + (v01 - v00) * tx;
  float v1 = v10 + (v11 - v10) * tx;
  dst[(size_t)bc * HW + y * 64 + x] = v0 + (v1 - v0) * ty;
}

// ---------------------------------------------------------------------------
// Bilinear upsample x2, half-pixel (align_corners=False): 32x32 -> 64x64
// ---------------------------------------------------------------------------
__global__ __launch_bounds__(256)
void k_resize_hp(const float* __restrict__ src, float* __restrict__ dst) {
  int i = blockIdx.x * 256 + threadIdx.x;
  int x = i & 63, y = (i >> 6) & 63;
  int bc = i >> 12;
  float fy = y * 0.5f - 0.25f;
  float fx = x * 0.5f - 0.25f;
  int y0 = (fy >= 0.f) ? (int)fy : -1;
  int x0 = (fx >= 0.f) ? (int)fx : -1;
  float ty = fy - (float)y0, tx = fx - (float)x0;
  int ya = y0 < 0 ? 0 : y0;           int xa = x0 < 0 ? 0 : x0;
  int yb = y0 + 1 > 31 ? 31 : y0 + 1; int xb = x0 + 1 > 31 ? 31 : x0 + 1;
  const float* p = src + (size_t)bc * 1024;
  float v00 = p[ya * 32 + xa], v01 = p[ya * 32 + xb];
  float v10 = p[yb * 32 + xa], v11 = p[yb * 32 + xb];
  float v0 = v00 + (v01 - v00) * tx;
  float v1 = v10 + (v11 - v10) * tx;
  dst[(size_t)bc * HW + y * 64 + x] = v0 + (v1 - v0) * ty;
}

// ---------------------------------------------------------------------------
// Depthwise 3x3, pad 1, NCHW f32 in -> NHWC f16 out ([b][pixel][dstC] GEMM layout).
// Block = 32 channels x 8 pixels, so NHWC stores are contiguous 64B runs per
// 32-lane group; NCHW reads hit each 64B line ~8x across the pixel tile.
// ---------------------------------------------------------------------------
__global__ __launch_bounds__(256)
void k_dw3x3(const float* __restrict__ src, const float* __restrict__ w9,
             half_t* __restrict__ dst, int Cin, int dstOff, int dstC) {
  int bid = blockIdx.x;
  int pt = bid & 511;              // HW/8 = 512 pixel tiles
  int rest = bid >> 9;
  int cT = Cin >> 5;               // channel tiles of 32
  int ct = rest % cT;
  int b  = rest / cT;
  int c = (ct << 5) + (threadIdx.x & 31);
  int p = (pt << 3) + (threadIdx.x >> 5);
  int x = p & 63, y = p >> 6;

  const float* sp = src + (size_t)(b * Cin + c) * HW;
  const float* wp = w9 + c * 9;
  float acc = 0.f;
#pragma unroll
  for (int dy = -1; dy <= 1; ++dy) {
    int yy = y + dy;
    if (yy < 0 || yy > 63) continue;
#pragma unroll
    for (int dx = -1; dx <= 1; ++dx) {
      int xx = x + dx;
      if (xx < 0 || xx > 63) continue;
      acc += sp[yy * 64 + xx] * wp[(dy + 1) * 3 + (dx + 1)];
    }
  }
  dst[((size_t)b * HW + p) * dstC + dstOff + c] = (half_t)acc;
}

// ---------------------------------------------------------------------------
// Batched per-sample GEMM via WMMA:  Y[b] = (Wbase + Dper[b]) [OxK] * X[b] [KxHW]
// X is NHWC f16 ([pixel][K]).  f16 operands, f32 accumulate.
// Block tile 64(M) x 256(N), K-chunks of 32, double-buffered LDS with
// async global->LDS b128 copies for the X tile (software pipeline:
// issue chunk i+1, s_wait_asynccnt for chunk i, barrier, 8 WMMAs/wave).
// 8 waves -> 2(M) x 4(N); each wave computes 32x64 = 2x4 v8f accumulators.
// Fragment lane maps used here (contiguous k order, same for A and B):
//   A 16x32:  lanes 0-15 M=l, K=0..15;       lanes 16-31 K=16..31
//   B 32x16:  lanes 0-15 N=l, K=0..15;       lanes 16-31 K=16..31
// With X stored [N][K] in LDS, each B-fragment is two contiguous b128 reads.
// ---------------------------------------------------------------------------
#define BM  64
#define BN  256
#define BK  32
#define LDA 36   // padded LDS row stride (halves) for W tile  (72B rows)
#define LDX 40   // padded LDS row stride (halves) for X tile  (80B rows, 16B-aligned)

__global__ __launch_bounds__(256)
void k_gemm_wmma(const float* __restrict__ Wbase, const float* __restrict__ Dper,
                 const half_t* __restrict__ X, float* __restrict__ Y,
                 int O, int K) {
  __shared__ __align__(16) half_t Wt[2][BM * LDA];
  __shared__ __align__(16) half_t Xl[2][BN * LDX];

  const int tid   = threadIdx.x;
  const int b     = blockIdx.z;
  const int mBase = blockIdx.y * BM;
  const int nBase = blockIdx.x * BN;

  const float*  Wp = Wbase;
  const float*  Dp = Dper + (size_t)b * O * K;
  const half_t* Xp = X + (size_t)b * HW * K;      // NHWC: [pixel][K]
  float*        Yp = Y + (size_t)b * O * HW;      // NCHW out

  const int wid   = tid >> 5;
  const int lane  = tid & 31;
  const int waveM = wid & 1;        // 2 waves along M
  const int waveN = wid >> 1;       // 4 waves along N (64 N each)
  const int lrow  = lane & 15;
  const bool hi   = lane >= 16;

  v8f acc[2][4] = {};

  // cooperative-load coords
  const int wm = tid >> 2, wk = (tid & 3) * 8;    // W tile: 64 rows x 32 K
  const int xn = tid;                              // X tile: one N-row per thread

  const float* wg = Wp + (size_t)(mBase + wm) * K + wk;
  const float* dg = Dp + (size_t)(mBase + wm) * K + wk;
  const half_t* xg = Xp + (size_t)(nBase + xn) * K;   // row of K halves

  const int nChunks = K / BK;

  // ---- stage one K-chunk into LDS buffer `buf` ----
  auto stage = [&](int kc, int buf) {
    // W tile: fp32 (pw + per-sample delta) -> f16, row-major [M][K]
    const float4* w4 = (const float4*)(wg + kc * BK);
    const float4* d4 = (const float4*)(dg + kc * BK);
    float4 a0 = w4[0], a1 = w4[1], e0 = d4[0], e1 = d4[1];
    half_t* o = &Wt[buf][wm * LDA + wk];
    o[0] = (half_t)(a0.x + e0.x); o[1] = (half_t)(a0.y + e0.y);
    o[2] = (half_t)(a0.z + e0.z); o[3] = (half_t)(a0.w + e0.w);
    o[4] = (half_t)(a1.x + e1.x); o[5] = (half_t)(a1.y + e1.y);
    o[6] = (half_t)(a1.z + e1.z); o[7] = (half_t)(a1.w + e1.w);
    // X tile: 4 x 16B async copies straight into LDS, [N][K] layout
    const half_t* gsrc = xg + kc * BK;
    half_t* ldst = &Xl[buf][xn * LDX];
#pragma unroll
    for (int j = 0; j < 4; ++j) cp16_g2l(ldst + j * 8, gsrc + j * 8);
  };

  stage(0, 0);
  for (int i = 0; i < nChunks; ++i) {
    const int cur = i & 1, nxt = cur ^ 1;
    if (i + 2 < nChunks)  // cache prefetch two chunks ahead (global_prefetch_b8)
      __builtin_prefetch(wg + (i + 2) * BK, 0, 1);
    if (i + 1 < nChunks) {
      stage(i + 1, nxt);
      wait_async_le<4>();   // my 4 older (chunk i) async copies have landed
    } else {
      wait_async_le<0>();
    }
    __syncthreads();

    // ---- build fragments and issue 2x4 WMMAs ----
    v16h afr[2], bfr[4];
#pragma unroll
    for (int mi = 0; mi < 2; ++mi) {
      const half_t* wr = &Wt[cur][(waveM * 32 + mi * 16 + lrow) * LDA + (hi ? 16 : 0)];
#pragma unroll
      for (int j = 0; j < 8; ++j) { afr[mi][j] = wr[j]; afr[mi][8 + j] = wr[8 + j]; }
    }
#pragma unroll
    for (int ni = 0; ni < 4; ++ni) {
      const half_t* xr = &Xl[cur][(waveN * 64 + ni * 16 + lrow) * LDX + (hi ? 16 : 0)];
#pragma unroll
      for (int j = 0; j < 16; ++j) bfr[ni][j] = xr[j];
    }
#pragma unroll
    for (int mi = 0; mi < 2; ++mi)
#pragma unroll
      for (int ni = 0; ni < 4; ++ni)
        acc[mi][ni] = __builtin_amdgcn_wmma_f32_16x16x32_f16(
            false, afr[mi], false, bfr[ni], (short)0, acc[mi][ni], false, false);
    __syncthreads();   // protect buffers before next stage overwrites
  }

  // ---- epilogue: C/D layout — VGPR r holds M=r (lanes 0-15) / M=8+r (16-31) ----
#pragma unroll
  for (int mi = 0; mi < 2; ++mi)
#pragma unroll
    for (int ni = 0; ni < 4; ++ni) {
      int m0 = mBase + waveM * 32 + mi * 16 + (hi ? 8 : 0);
      int n0 = nBase + waveN * 64 + ni * 16 + lrow;
#pragma unroll
      for (int r = 0; r < 8; ++r)
        Yp[(size_t)(m0 + r) * HW + n0] = acc[mi][ni][r];
    }
}

// ---------------------------------------------------------------------------
// GroupNorm stats: one block per (b, group); 32 groups, C/32 channels each.
// ---------------------------------------------------------------------------
__global__ __launch_bounds__(256)
void k_gnstats(const float* __restrict__ src, float* __restrict__ stats, int C) {
  const int bg = blockIdx.x;
  const int b = bg >> 5, g = bg & 31;
  const int cpg = C >> 5;
  const int n = cpg * HW;
  const float* p = src + ((size_t)b * C + (size_t)g * cpg) * HW;
  float s = 0.f, q = 0.f;
  for (int i = threadIdx.x; i < n; i += 256) { float v = p[i]; s += v; q += v * v; }
  __shared__ float rs[256], rq[256];
  rs[threadIdx.x] = s; rq[threadIdx.x] = q;
  __syncthreads();
  for (int off = 128; off > 0; off >>= 1) {
    if (threadIdx.x < off) {
      rs[threadIdx.x] += rs[threadIdx.x + off];
      rq[threadIdx.x] += rq[threadIdx.x + off];
    }
    __syncthreads();
  }
  if (threadIdx.x == 0) {
    float mean = rs[0] / (float)n;
    float var  = rq[0] / (float)n - mean * mean;
    stats[bg * 2]     = mean;
    stats[bg * 2 + 1] = rsqrtf(var + 1e-5f);
  }
}

// ---------------------------------------------------------------------------
// GroupNorm apply (scale/shift) + ReLU
// ---------------------------------------------------------------------------
__global__ __launch_bounds__(256)
void k_gnapply(const float* __restrict__ src, const float* __restrict__ stats,
               const float* __restrict__ gamma, const float* __restrict__ beta,
               float* __restrict__ dst, int C) {
  size_t i = (size_t)blockIdx.x * 256 + threadIdx.x;  // over B*C*HW
  int bc = (int)(i >> 12);
  int c = bc % C, b = bc / C;
  int g = c / (C >> 5);
  float mean = stats[(b * 32 + g) * 2];
  float rstd = stats[(b * 32 + g) * 2 + 1];
  float v = (src[i] - mean) * rstd * gamma[c] + beta[c];
  dst[i] = v > 0.f ? v : 0.f;
}

// ---------------------------------------------------------------------------
// Host orchestration
// ---------------------------------------------------------------------------
extern "C" void kernel_launch(void* const* d_in, const int* in_sizes, int n_in,
                              void* d_out, int out_size, void* d_ws, size_t ws_size,
                              hipStream_t stream) {
  const float* x1      = (const float*)d_in[0];
  const float* x2      = (const float*)d_in[1];
  const float* cx1_dw1 = (const float*)d_in[2];
  const float* cx1_pw1 = (const float*)d_in[3];
  const float* cx1_g1  = (const float*)d_in[4];
  const float* cx1_b1  = (const float*)d_in[5];
  const float* cx1_dw2 = (const float*)d_in[6];
  const float* cx1_pw2 = (const float*)d_in[7];
  const float* cx1_g2  = (const float*)d_in[8];
  const float* cx1_b2  = (const float*)d_in[9];
  const float* cx2_dw1 = (const float*)d_in[10];
  const float* cx2_pw1 = (const float*)d_in[11];
  const float* cx2_g1  = (const float*)d_in[12];
  const float* cx2_b1  = (const float*)d_in[13];
  const float* cx2_dw2 = (const float*)d_in[14];
  const float* cx2_pw2 = (const float*)d_in[15];
  const float* cx2_g2  = (const float*)d_in[16];
  const float* cx2_b2  = (const float*)d_in[17];
  const float* cf_dw1  = (const float*)d_in[18];
  const float* cf_pw1  = (const float*)d_in[19];
  const float* cf_g1   = (const float*)d_in[20];
  const float* cf_b1   = (const float*)d_in[21];
  const float* cf_dw2  = (const float*)d_in[22];
  const float* cf_pw2  = (const float*)d_in[23];
  const float* cf_g2   = (const float*)d_in[24];
  const float* cf_b2   = (const float*)d_in[25];
  const float* d_cx1_1 = (const float*)d_in[26];
  const float* d_cx1_2 = (const float*)d_in[27];
  const float* d_cx2_1 = (const float*)d_in[28];
  const float* d_cx2_2 = (const float*)d_in[29];
  const float* d_cf_1  = (const float*)d_in[30];
  const float* d_cf_2  = (const float*)d_in[31];

  // Workspace layout (192 MiB + stats):
  char* ws = (char*)d_ws;
  float*  bufA  = (float*)(ws);                        // 64 MiB: up1; later GEMM out (G)
  float*  bufB  = (float*)(ws + (size_t)(64u << 20));  // 64 MiB: up2; later x2_p
  float*  bufP1 = (float*)(ws + (size_t)(128u << 20)); // 32 MiB: x1_p
  half_t* bufH  = (half_t*)(ws + (size_t)(160u << 20));// 32 MiB: f16 NHWC GEMM operand
  float*  stats = (float*)(ws + (size_t)(192u << 20)); // 2 KiB: GN mean/rstd
  float*  bufG  = bufA;
  float*  bufP2 = bufB;

  const dim3 blk(256);
  const int n512 = BATCH * 512 * HW;  // 16.7M elems
  const int n256 = BATCH * 256 * HW;  // 8.4M elems

  auto gemm = [&](const float* Wb, const float* Dp, const half_t* X, float* Y, int K) {
    dim3 g(HW / BN, 256 / BM, BATCH);   // (16, 4, 8)
    k_gemm_wmma<<<g, blk, 0, stream>>>(Wb, Dp, X, Y, 256, K);
  };
  auto gn = [&](const float* src, const float* gamma, const float* beta, float* dst) {
    k_gnstats<<<BATCH * 32, blk, 0, stream>>>(src, stats, 256);
    k_gnapply<<<n256 / 256, blk, 0, stream>>>(src, stats, gamma, beta, dst, 256);
  };

  // Upsample both inputs
  k_upsample_ac<<<n512 / 256, blk, 0, stream>>>(x1, bufA);
  k_resize_hp  <<<n512 / 256, blk, 0, stream>>>(x2, bufB);

  // Branch cx1
  k_dw3x3<<<n512 / 256, blk, 0, stream>>>(bufA, cx1_dw1, bufH, 512, 0, 512);
  gemm(cx1_pw1, d_cx1_1, bufH, bufG, 512);
  gn(bufG, cx1_g1, cx1_b1, bufG);
  k_dw3x3<<<n256 / 256, blk, 0, stream>>>(bufG, cx1_dw2, bufH, 256, 0, 256);
  gemm(cx1_pw2, d_cx1_2, bufH, bufP1, 256);
  gn(bufP1, cx1_g2, cx1_b2, bufP1);

  // Branch cx2
  k_dw3x3<<<n512 / 256, blk, 0, stream>>>(bufB, cx2_dw1, bufH, 512, 0, 512);
  gemm(cx2_pw1, d_cx2_1, bufH, bufG, 512);
  gn(bufG, cx2_g1, cx2_b1, bufG);
  k_dw3x3<<<n256 / 256, blk, 0, stream>>>(bufG, cx2_dw2, bufH, 256, 0, 256);
  gemm(cx2_pw2, d_cx2_2, bufH, bufP2, 256);
  gn(bufP2, cx2_g2, cx2_b2, bufP2);

  // Fuse stage (concat is free: depthwise writes channel-offset slabs of bufH)
  k_dw3x3<<<n256 / 256, blk, 0, stream>>>(bufP1, cf_dw1,           bufH, 256, 0,   512);
  k_dw3x3<<<n256 / 256, blk, 0, stream>>>(bufP2, cf_dw1 + 256 * 9, bufH, 256, 256, 512);
  gemm(cf_pw1, d_cf_1, bufH, bufG, 512);
  gn(bufG, cf_g1, cf_b1, bufG);
  k_dw3x3<<<n256 / 256, blk, 0, stream>>>(bufG, cf_dw2, bufH, 256, 0, 256);
  gemm(cf_pw2, d_cf_2, bufH, bufG, 256);
  gn(bufG, cf_g2, cf_b2, (float*)d_out);
}
